// Attention_48661979463892
// MI455X (gfx1250) — matrix operand order repaired and checked by reading, not compiled, not run on hardware
//
#include <hip/hip_runtime.h>

typedef unsigned short u16;
typedef __attribute__((ext_vector_type(16))) __bf16        v16bf;
typedef __attribute__((ext_vector_type(8)))  float         v8f;
typedef __attribute__((ext_vector_type(4)))  unsigned int  v4u;

union FragU { v16bf f; v4u u[2]; };

// ---------- helpers ----------

__device__ __forceinline__ u16 f2bf(float x) {
  unsigned u = __builtin_bit_cast(unsigned, x);
  u += 0x7FFFu + ((u >> 16) & 1u);   // round-to-nearest-even
  return (u16)(u >> 16);
}

// A fragment (16x32, M x K), source row-major [M][K] with leading dim ld.
// Per ISA: lanes 0-15 hold M=m, K={0..7,16..23}; lanes 16-31 K={8..15,24..31}.
__device__ __forceinline__ v16bf load_frag_a(const u16* p, int ld) {
  const int lane = threadIdx.x & 31;
  const int m = lane & 15, hi = lane >> 4;
  const u16* r = p + (size_t)m * ld + hi * 8;
  FragU f;
  f.u[0] = *(const v4u*)(r);
  f.u[1] = *(const v4u*)(r + 16);
  return f.f;
}

// B fragment (32x16, K x N), source row-major [N][K] with leading dim ld.
// Per ISA: lanes 0-15 hold N=n, K={0..7,16..23}; lanes 16-31 K={8..15,24..31}.
__device__ __forceinline__ v16bf load_frag_b(const u16* p, int ld) {
  const int lane = threadIdx.x & 31;
  const int n = lane & 15, hi = lane >> 4;
  const u16* r = p + (size_t)n * ld + hi * 8;
  FragU f;
  f.u[0] = *(const v4u*)(r);
  f.u[1] = *(const v4u*)(r + 16);
  return f.f;
}

__device__ __forceinline__ v8f wmma_bf16(v16bf a, v16bf b, v8f c) {
  return __builtin_amdgcn_wmma_f32_16x16x32_bf16(false, a, false, b, (short)0, c,
                                                 false, false);
}

__device__ __forceinline__ v8f zero8() {
  v8f z = {0.f, 0.f, 0.f, 0.f, 0.f, 0.f, 0.f, 0.f};
  return z;
}

// CDNA5 async copy: global -> LDS, 16 bytes per enabled lane, tracked by ASYNCcnt.
// LDS destination address = low 32 bits of the generic shared pointer.
__device__ __forceinline__ void async_ld_b128(void* lds_ptr, const void* gptr) {
  unsigned lds_off = (unsigned)(uintptr_t)lds_ptr;
  unsigned long long ga = (unsigned long long)(uintptr_t)gptr;
  asm volatile("global_load_async_to_lds_b128 %0, %1, off"
               :: "v"(lds_off), "v"(ga) : "memory");
}

__device__ __forceinline__ void wait_async0() {
  asm volatile("s_wait_asynccnt 0x0" ::: "memory");
}

// ---------- stage 0: casts ----------

__global__ void cast_bf16_kernel(const float* __restrict__ src,
                                 u16* __restrict__ dst, int n) {
  int i = blockIdx.x * blockDim.x + threadIdx.x;
  if (i < n) dst[i] = f2bf(src[i]);
}

// src f32 [b][C][L] -> dst bf16 [b][L][C]
__global__ void transpose_cast_kernel(const float* __restrict__ src,
                                      u16* __restrict__ dst, int C, int L) {
  __shared__ float tile[32][33];
  const float* s = src + (size_t)blockIdx.z * C * L;
  u16* d = dst + (size_t)blockIdx.z * C * L;
  const int t0 = blockIdx.x * 32, c0 = blockIdx.y * 32;
  const int tx = threadIdx.x, ty = threadIdx.y;  // block (32,8)
#pragma unroll
  for (int j = 0; j < 4; ++j)
    tile[ty + 8 * j][tx] = s[(size_t)(c0 + ty + 8 * j) * L + t0 + tx];
  __syncthreads();
#pragma unroll
  for (int j = 0; j < 4; ++j)
    d[(size_t)(t0 + ty + 8 * j) * C + c0 + tx] = f2bf(tile[tx][ty + 8 * j]);
}

// ---------- stage 1: projection GEMMs ----------
// C[m][n] = sum_k A[m][k] * Bm[b][n][k]; bf16 out with head-split layouts:
// mode 0: out[(bh*N + n)*64 + (m%64)]   ([bh][t][k]  for K and Q)
// mode 1: out[(bh*64 + (m%64))*N + n]   ([bh][v][t]  for V)
__global__ void gemm_proj_kernel(const u16* __restrict__ A,
                                 const u16* __restrict__ Bm,
                                 u16* __restrict__ out,
                                 int M, int N, int Kd, int mode) {
  const int wave = threadIdx.x >> 5;
  const int lane = threadIdx.x & 31;
  const int n_lo = lane & 15, hi = lane >> 4;
  const int m0 = blockIdx.y * 64;
  const int n0 = blockIdx.x * 128 + wave * 32;
  const u16* Bb = Bm + (size_t)blockIdx.z * N * Kd;

  v8f acc[4][2];
#pragma unroll
  for (int i = 0; i < 4; ++i)
#pragma unroll
    for (int j = 0; j < 2; ++j) acc[i][j] = zero8();

  for (int kk = 0; kk < Kd; kk += 32) {
    v16bf a[4], b[2];
#pragma unroll
    for (int i = 0; i < 4; ++i)
      a[i] = load_frag_a(A + (size_t)(m0 + 16 * i) * Kd + kk, Kd);
#pragma unroll
    for (int j = 0; j < 2; ++j)
      b[j] = load_frag_b(Bb + (size_t)(n0 + 16 * j) * Kd + kk, Kd);
#pragma unroll
    for (int i = 0; i < 4; ++i)
#pragma unroll
      for (int j = 0; j < 2; ++j) acc[i][j] = wmma_bf16(a[i], b[j], acc[i][j]);
  }

#pragma unroll
  for (int i = 0; i < 4; ++i)
#pragma unroll
    for (int j = 0; j < 2; ++j)
#pragma unroll
      for (int r = 0; r < 8; ++r) {
        int m = m0 + 16 * i + r + 8 * hi;
        int n = n0 + 16 * j + n_lo;
        int h = m >> 6, c = m & 63;
        int bh = blockIdx.z * 8 + h;
        size_t idx = (mode == 0) ? ((size_t)bh * N + n) * 64 + c
                                 : ((size_t)bh * 64 + c) * N + n;
        out[idx] = f2bf(acc[i][j][r]);
      }
}

// ---------- stage 2: fused flash attention ----------
// Kt [bh][T][64], Vt [bh][64][T], Qs [bh][S][64], mask f32 [T][S]
// attb bf16 [b][S][512]. Softmax over t per query column s (axis=1 in ref).
__global__ void flash_attn_kernel(const u16* __restrict__ Kt,
                                  const u16* __restrict__ Vt,
                                  const u16* __restrict__ Qs,
                                  const float* __restrict__ mask,
                                  u16* __restrict__ attb, int T, int S) {
  __shared__ u16 Ks[64][72];       // [t_local][k]
  __shared__ u16 Vs[64][72];       // [v][t_local]
  __shared__ u16 Ps[4][16][72];    // per wave: [s_local][t_local]
  __shared__ float Ms[64][68];     // mask tile [t_local][s_local]

  const int tid = threadIdx.x;
  const int wave = tid >> 5, lane = tid & 31;
  const int n_lo = lane & 15, hi = lane >> 4;
  const int bh = blockIdx.y;
  const int s0 = blockIdx.x * 64;
  const int sw = s0 + wave * 16;

  const u16* Kb = Kt + (size_t)bh * T * 64;
  const u16* Vb = Vt + (size_t)bh * 64 * T;
  const u16* Qb = Qs + (size_t)bh * S * 64;

  v16bf qf[2];
#pragma unroll
  for (int k2 = 0; k2 < 2; ++k2)
    qf[k2] = load_frag_b(Qb + (size_t)sw * 64 + k2 * 32, 64);

  v8f o_acc[4];
#pragma unroll
  for (int i = 0; i < 4; ++i) o_acc[i] = zero8();
  float m_run = -3.0e38f, l_run = 0.f;

  for (int t0 = 0; t0 < T; t0 += 64) {
    __syncthreads();  // all waves done reading previous tiles
    // K/V bf16 tiles: 64 rows x 8 x (8 bf16) chunks, async global->LDS
    for (int idx = tid; idx < 64 * 8; idx += 128) {
      int row = idx >> 3, c = (idx & 7) * 8;
      async_ld_b128(&Ks[row][c], Kb + (size_t)(t0 + row) * 64 + c);
      async_ld_b128(&Vs[row][c], Vb + (size_t)row * T + t0 + c);
    }
    // mask f32 tile: 64 rows x 16 x (4 f32) chunks
    for (int idx = tid; idx < 64 * 16; idx += 128) {
      int row = idx >> 4, c = (idx & 15) * 4;
      async_ld_b128(&Ms[row][c], mask + (size_t)(t0 + row) * S + s0 + c);
    }
    wait_async0();
    __syncthreads();

    // scores: D[m=t_local][n=s_local], plus mask and 1/sqrt(dk) scaling
    v8f sc[4];
    float pm = -3.0e38f;
#pragma unroll
    for (int mt = 0; mt < 4; ++mt) {
      v8f s = zero8();
#pragma unroll
      for (int k2 = 0; k2 < 2; ++k2) {
        v16bf a = load_frag_a(&Ks[mt * 16][k2 * 32], 72);
        s = wmma_bf16(a, qf[k2], s);
      }
#pragma unroll
      for (int r = 0; r < 8; ++r) {
        int tl = mt * 16 + r + 8 * hi;
        float v = 0.125f * (s[r] + Ms[tl][wave * 16 + n_lo]);
        s[r] = v;
        pm = fmaxf(pm, v);
      }
      sc[mt] = s;
    }
    pm = fmaxf(pm, __shfl_xor(pm, 16, 32));  // combine the two M-halves
    float m_new = fmaxf(m_run, pm);
    float corr = __expf(m_run - m_new);
    l_run *= corr;
#pragma unroll
    for (int mv = 0; mv < 4; ++mv)
#pragma unroll
      for (int r = 0; r < 8; ++r) o_acc[mv][r] *= corr;

    float psum = 0.f;
#pragma unroll
    for (int mt = 0; mt < 4; ++mt)
#pragma unroll
      for (int r = 0; r < 8; ++r) {
        float p = __expf(sc[mt][r] - m_new);
        psum += p;
        Ps[wave][n_lo][mt * 16 + r + 8 * hi] = f2bf(p);  // transpose to [s][t]
      }
    psum += __shfl_xor(psum, 16, 32);
    l_run += psum;
    m_run = m_new;

    // O[v][s] += V[v][t] * P[t][s]  (intra-wave DS ops are in-order)
#pragma unroll
    for (int mv = 0; mv < 4; ++mv)
#pragma unroll
      for (int kt = 0; kt < 2; ++kt) {
        v16bf a = load_frag_a(&Vs[mv * 16][kt * 32], 72);
        v16bf b = load_frag_b(&Ps[wave][0][kt * 32], 72);
        o_acc[mv] = wmma_bf16(a, b, o_acc[mv]);
      }
  }

  const float inv = 1.f / l_run;
  const int b = bh >> 3, h = bh & 7;
#pragma unroll
  for (int mv = 0; mv < 4; ++mv)
#pragma unroll
    for (int r = 0; r < 8; ++r) {
      int v = mv * 16 + r + 8 * hi;
      int sg = sw + n_lo;
      attb[((size_t)b * S + sg) * 512 + h * 64 + v] = f2bf(o_acc[mv][r] * inv);
    }
}

// ---------- stage 3: final linear ----------
// out[b][s][o] = sum_v attb[b][s][v] * W[o][v] + bias[o]
__global__ void gemm_final_kernel(const u16* __restrict__ A,
                                  const u16* __restrict__ Bw,
                                  const float* __restrict__ bias,
                                  float* __restrict__ out,
                                  int Mdim, int Ndim, int Kd) {
  const int wave = threadIdx.x >> 5;
  const int lane = threadIdx.x & 31;
  const int n_lo = lane & 15, hi = lane >> 4;
  const int m0 = blockIdx.y * 64;
  const int n0 = blockIdx.x * 128 + wave * 32;
  const u16* Ab = A + (size_t)blockIdx.z * Mdim * Kd;

  v8f acc[4][2];
#pragma unroll
  for (int i = 0; i < 4; ++i)
#pragma unroll
    for (int j = 0; j < 2; ++j) acc[i][j] = zero8();

  for (int kk = 0; kk < Kd; kk += 32) {
    v16bf a[4], b[2];
#pragma unroll
    for (int i = 0; i < 4; ++i)
      a[i] = load_frag_a(Ab + (size_t)(m0 + 16 * i) * Kd + kk, Kd);
#pragma unroll
    for (int j = 0; j < 2; ++j)
      b[j] = load_frag_b(Bw + (size_t)(n0 + 16 * j) * Kd + kk, Kd);
#pragma unroll
    for (int i = 0; i < 4; ++i)
#pragma unroll
      for (int j = 0; j < 2; ++j) acc[i][j] = wmma_bf16(a[i], b[j], acc[i][j]);
  }

#pragma unroll
  for (int i = 0; i < 4; ++i)
#pragma unroll
    for (int j = 0; j < 2; ++j)
#pragma unroll
      for (int r = 0; r < 8; ++r) {
        int m = m0 + 16 * i + r + 8 * hi;
        int n = n0 + 16 * j + n_lo;
        out[((size_t)blockIdx.z * Mdim + m) * Ndim + n] = acc[i][j][r] + bias[n];
      }
}

// ---------- launch ----------

extern "C" void kernel_launch(void* const* d_in, const int* in_sizes, int n_in,
                              void* d_out, int out_size, void* d_ws, size_t ws_size,
                              hipStream_t stream) {
  (void)in_sizes; (void)n_in; (void)out_size; (void)ws_size;
  const float* x    = (const float*)d_in[0];
  const float* y    = (const float*)d_in[1];
  const float* mask = (const float*)d_in[2];
  const float* Wk   = (const float*)d_in[3];
  const float* Wv   = (const float*)d_in[4];
  const float* Wq   = (const float*)d_in[5];
  const float* W    = (const float*)d_in[6];
  const float* bias = (const float*)d_in[7];
  float* out = (float*)d_out;

  const int Bn = 2, I = 512, O = 512, T = 2048, S = 2048, E = 512;

  char* ws = (char*)d_ws;
  size_t off = 0;
  auto alloc = [&](size_t bytes) {
    void* p = ws + off;
    off += (bytes + 255) & ~(size_t)255;
    return p;
  };
  u16* xT   = (u16*)alloc((size_t)Bn * T * I * 2);   // [b][t][i]
  u16* yT   = (u16*)alloc((size_t)Bn * S * O * 2);   // [b][s][o]
  u16* Wkb  = (u16*)alloc((size_t)E * I * 2);
  u16* Wvb  = (u16*)alloc((size_t)E * I * 2);
  u16* Wqb  = (u16*)alloc((size_t)E * O * 2);
  u16* Wb   = (u16*)alloc((size_t)O * E * 2);
  u16* Kt   = (u16*)alloc((size_t)16 * T * 64 * 2);  // [bh][t][k]
  u16* Qsb  = (u16*)alloc((size_t)16 * S * 64 * 2);  // [bh][s][k]
  u16* Vt   = (u16*)alloc((size_t)16 * 64 * T * 2);  // [bh][v][t]
  u16* attb = (u16*)alloc((size_t)Bn * S * 512 * 2); // [b][s][h*64+v]

  cast_bf16_kernel<<<(E * I + 255) / 256, 256, 0, stream>>>(Wk, Wkb, E * I);
  cast_bf16_kernel<<<(E * I + 255) / 256, 256, 0, stream>>>(Wv, Wvb, E * I);
  cast_bf16_kernel<<<(E * O + 255) / 256, 256, 0, stream>>>(Wq, Wqb, E * O);
  cast_bf16_kernel<<<(O * E + 255) / 256, 256, 0, stream>>>(W, Wb, O * E);

  transpose_cast_kernel<<<dim3(T / 32, I / 32, Bn), dim3(32, 8), 0, stream>>>(x, xT, I, T);
  transpose_cast_kernel<<<dim3(S / 32, O / 32, Bn), dim3(32, 8), 0, stream>>>(y, yT, O, S);

  gemm_proj_kernel<<<dim3(T / 128, E / 64, Bn), 128, 0, stream>>>(Wkb, xT, Kt, E, T, I, 0);
  gemm_proj_kernel<<<dim3(S / 128, E / 64, Bn), 128, 0, stream>>>(Wqb, yT, Qsb, E, S, O, 0);
  gemm_proj_kernel<<<dim3(T / 128, E / 64, Bn), 128, 0, stream>>>(Wvb, xT, Vt, E, T, I, 1);

  flash_attn_kernel<<<dim3(S / 64, 16), 128, 0, stream>>>(Kt, Vt, Qsb, mask, attb, T, S);

  gemm_final_kernel<<<dim3(O / 128, S / 64, Bn), 128, 0, stream>>>(attb, Wb, bias, out, S, O, E);
}
